// Seq2SeqLSTM_63410897158894
// MI455X (gfx1250) — matrix operand order repaired and checked by reading, not compiled, not run on hardware
//
#include <hip/hip_runtime.h>

// ---------------------------------------------------------------------------
// Seq2Seq LSTM (enc 2xLSTM over 168 steps, dec 24 autoregressive steps)
// One workgroup owns a 32-row batch tile for the whole sequence (recurrence is
// batch-independent -> no inter-WG sync). Weights staged f32->f16 into LDS
// once, then each wave hoists its 14 invariant B-fragments (112 VGPRs) into
// registers for the entire phase. Gate GEMMs via V_WMMA_F32_16X16X32_F16;
// cell nonlinearity in fp32 VALU with v_rcp/v_exp fast paths.
// ---------------------------------------------------------------------------

typedef __attribute__((ext_vector_type(16))) _Float16 v16h;
typedef __attribute__((ext_vector_type(8)))  _Float16 v8h;
typedef __attribute__((ext_vector_type(8)))  float    v8f;

#define BATCH    4096
#define LOOKBACK 168
#define HORIZON  24
#define NFEAT    32
#define NTGT     8
#define HID      64
#define G4       256      // 4*HID gate width
#define BTILE    32       // batch rows per workgroup
#define NTHREADS 256      // 8 waves (wave32)

// ---- LDS layout (bytes, all 64B aligned) ----------------------------------
constexpr int S_WIH0  = 0;                         // [256][32] f16 (enc/dec l0 input W, dec padded 8->32)
constexpr int S_WHH0  = S_WIH0  + G4*NFEAT*2;      // [256][64] f16
constexpr int S_WIH1  = S_WHH0  + G4*HID*2;        // [256][64] f16
constexpr int S_WHH1  = S_WIH1  + G4*HID*2;        // [256][64] f16
constexpr int S_FCW   = S_WHH1  + G4*HID*2;        // [8][64]   f16
constexpr int S_B0    = S_FCW   + NTGT*HID*2;      // [256]     f32 (bih+bhh layer0)
constexpr int S_B1    = S_B0    + G4*4;            // [256]     f32 (layer1)
constexpr int S_FCB   = S_B1    + G4*4;            // [8]       f32 (pad to 64B)
constexpr int S_GATES = S_FCB   + 64;              // [32][256] f32
constexpr int S_H0    = S_GATES + BTILE*G4*4;      // [32][64]  f16
constexpr int S_H1    = S_H0    + BTILE*HID*2;     // [32][64]  f16
constexpr int S_C0    = S_H1    + BTILE*HID*2;     // [32][64]  f32
constexpr int S_C1    = S_C0    + BTILE*HID*4;     // [32][64]  f32
constexpr int S_DIN   = S_C1    + BTILE*HID*4;     // [32][32]  f16 (x_t stage / dec feedback, K-padded)
constexpr int S_TOTAL = S_DIN   + BTILE*NFEAT*2;   // = 177216 bytes < 320KB/WGP

// ---------------------------------------------------------------------------
__device__ __forceinline__ float frcp(float x) { return __builtin_amdgcn_rcpf(x); }
__device__ __forceinline__ float sigm(float x) { return frcp(1.0f + __expf(-x)); }
__device__ __forceinline__ float ftanh(float x) {
    float e = __expf(-2.0f * x);
    return (1.0f - e) * frcp(1.0f + e);
}

__device__ __forceinline__ v8f wmma32(v16h a, v16h b, v8f c) {
    // D = A(16x32 f16) * B(32x16 f16) + C(16x16 f32)
    return __builtin_amdgcn_wmma_f32_16x16x32_f16(
        /*neg_a=*/false, a, /*neg_b=*/false, b,
        /*c_mod=*/(short)0, c, /*reuse_a=*/false, /*reuse_b=*/false);
}

// A fragment (16x32, f16) from an LDS row-major [rows][ld] buffer.
// ISA layout: lane (m = lane&15, half = lane>>4); elem e: K = 16*(e>=8) + 8*half + (e&7)
__device__ __forceinline__ v16h a_frag(const _Float16* Abuf, int ld, int m0, int kb, int lane) {
    int m  = m0 + (lane & 15);
    int hf = lane >> 4;
    const _Float16* p = Abuf + m * ld + kb + 8 * hf;
    v8h lo = *(const v8h*)(p);        // K = kb + 8*hf + [0..7]
    v8h hi = *(const v8h*)(p + 16);   // K = kb + 16 + 8*hf + [0..7]
    return __builtin_shufflevector(lo, hi, 0,1,2,3,4,5,6,7,8,9,10,11,12,13,14,15);
}

// B fragment (32x16, f16) for gates = A @ W^T : B[k][n] = W[n][k], W row-major [256][ld].
// ISA layout: lane (n = lane&15, half = lane>>4); elem e: K = kb + 16*(e>=8) + 8*half + (e&7)
__device__ __forceinline__ v16h b_frag(const _Float16* W, int ld, int n0, int kb, int lane) {
    int n  = n0 + (lane & 15);
    int hf = lane >> 4;
    const _Float16* p = W + n * ld + kb + 8 * hf;
    v8h lo = *(const v8h*)(p);
    v8h hi = *(const v8h*)(p + 16);
    return __builtin_shufflevector(lo, hi, 0,1,2,3,4,5,6,7,8,9,10,11,12,13,14,15);
}

// Per-wave register-resident weights: 14 x v16h = 112 VGPRs + 4 bias scalars.
struct WaveW {
    v16h wI0[2];      // Wih0 (K=32), per N-tile
    v16h wH0[2][2];   // Whh0 (K=64), per N-tile x K-chunk
    v16h wI1[2][2];   // Wih1 (K=64)
    v16h wH1[2][2];   // Whh1 (K=64)
    float b0[2], b1[2];
};

__device__ __forceinline__ void load_wave_weights(
    WaveW& W, const _Float16* sWih0, const _Float16* sWhh0,
    const _Float16* sWih1, const _Float16* sWhh1,
    const float* sB0, const float* sB1, int lane, int wave)
{
#pragma unroll
    for (int nt = 0; nt < 2; ++nt) {
        const int n0 = wave * 32 + nt * 16;
        W.wI0[nt]    = b_frag(sWih0, NFEAT, n0, 0,  lane);
        W.wH0[nt][0] = b_frag(sWhh0, HID,   n0, 0,  lane);
        W.wH0[nt][1] = b_frag(sWhh0, HID,   n0, 32, lane);
        W.wI1[nt][0] = b_frag(sWih1, HID,   n0, 0,  lane);
        W.wI1[nt][1] = b_frag(sWih1, HID,   n0, 32, lane);
        W.wH1[nt][0] = b_frag(sWhh1, HID,   n0, 0,  lane);
        W.wH1[nt][1] = b_frag(sWhh1, HID,   n0, 32, lane);
        W.b0[nt] = sB0[n0 + (lane & 15)];
        W.b1[nt] = sB1[n0 + (lane & 15)];
    }
}

__device__ __forceinline__ void store_tile(float* gates, const v8f& acc,
                                           int m0, int n0, int lane) {
    // D layout: VGPR r -> row m0 + r + 8*half, col n0 + (lane&15)
    const int hf = lane >> 4;
    const int nn = n0 + (lane & 15);
#pragma unroll
    for (int r = 0; r < 8; ++r)
        gates[(m0 + r + 8 * hf) * G4 + nn] = acc[r];
}

// layer0: gates = b0 + Din[32][32] @ Wih0^T + H0[32][64] @ Whh0^T
__device__ __forceinline__ void gemm_l0(const WaveW& W, const _Float16* sDin,
                                        const _Float16* sH0, float* gates,
                                        int lane, int wave)
{
#pragma unroll
    for (int mt = 0; mt < 2; ++mt) {
        const int m0 = mt * 16;
        v16h aI  = a_frag(sDin, NFEAT, m0, 0,  lane);
        v16h aH0 = a_frag(sH0,  HID,   m0, 0,  lane);
        v16h aH1 = a_frag(sH0,  HID,   m0, 32, lane);
#pragma unroll
        for (int nt = 0; nt < 2; ++nt) {
            v8f acc;
#pragma unroll
            for (int r = 0; r < 8; ++r) acc[r] = W.b0[nt];
            acc = wmma32(aI,  W.wI0[nt],    acc);
            acc = wmma32(aH0, W.wH0[nt][0], acc);
            acc = wmma32(aH1, W.wH0[nt][1], acc);
            store_tile(gates, acc, m0, wave * 32 + nt * 16, lane);
        }
    }
}

// layer1: gates = b1 + H0[32][64] @ Wih1^T + H1[32][64] @ Whh1^T
__device__ __forceinline__ void gemm_l1(const WaveW& W, const _Float16* sH0,
                                        const _Float16* sH1, float* gates,
                                        int lane, int wave)
{
#pragma unroll
    for (int mt = 0; mt < 2; ++mt) {
        const int m0 = mt * 16;
        v16h aI0 = a_frag(sH0, HID, m0, 0,  lane);
        v16h aI1 = a_frag(sH0, HID, m0, 32, lane);
        v16h aH0 = a_frag(sH1, HID, m0, 0,  lane);
        v16h aH1 = a_frag(sH1, HID, m0, 32, lane);
#pragma unroll
        for (int nt = 0; nt < 2; ++nt) {
            v8f acc;
#pragma unroll
            for (int r = 0; r < 8; ++r) acc[r] = W.b1[nt];
            acc = wmma32(aI0, W.wI1[nt][0], acc);
            acc = wmma32(aI1, W.wI1[nt][1], acc);
            acc = wmma32(aH0, W.wH1[nt][0], acc);
            acc = wmma32(aH1, W.wH1[nt][1], acc);
            store_tile(gates, acc, m0, wave * 32 + nt * 16, lane);
        }
    }
}

// LSTM cell elementwise (PyTorch gate order i,f,g,o). 2048 elems / 256 thr.
__device__ __forceinline__ void cell_update(
    const float* __restrict__ gates, _Float16* __restrict__ h,
    float* __restrict__ c, int tid)
{
#pragma unroll
    for (int e = 0; e < 8; ++e) {
        const int idx = tid + e * NTHREADS;          // = row*64 + col
        const int row = idx >> 6, col = idx & 63;
        const float* gr = gates + row * G4;
        const float gi = gr[col], gf = gr[64 + col], gg = gr[128 + col], go = gr[192 + col];
        float cv = c[idx];
        cv = sigm(gf) * cv + sigm(gi) * ftanh(gg);
        c[idx] = cv;
        h[idx] = (_Float16)(sigm(go) * ftanh(cv));
    }
}

__device__ __forceinline__ void cvt_copy(_Float16* dst, const float* src, int n, int tid) {
    for (int i = tid; i < n; i += NTHREADS) dst[i] = (_Float16)src[i];
}
// pad K: src [rows][kin] -> dst [rows][kout], zeros beyond kin
__device__ __forceinline__ void cvt_pad(_Float16* dst, const float* src,
                                        int rows, int kin, int kout, int tid) {
    for (int i = tid; i < rows * kout; i += NTHREADS) {
        const int r = i / kout, k = i % kout;
        dst[i] = (k < kin) ? (_Float16)src[r * kin + k] : (_Float16)0.0f;
    }
}

// ---------------------------------------------------------------------------
__global__ __launch_bounds__(NTHREADS)
void seq2seq_lstm_kernel(
    const float* __restrict__ x,
    const float* __restrict__ eWih0, const float* __restrict__ eWhh0,
    const float* __restrict__ ebih0, const float* __restrict__ ebhh0,
    const float* __restrict__ eWih1, const float* __restrict__ eWhh1,
    const float* __restrict__ ebih1, const float* __restrict__ ebhh1,
    const float* __restrict__ dWih0, const float* __restrict__ dWhh0,
    const float* __restrict__ dbih0, const float* __restrict__ dbhh0,
    const float* __restrict__ dWih1, const float* __restrict__ dWhh1,
    const float* __restrict__ dbih1, const float* __restrict__ dbhh1,
    const float* __restrict__ fcW, const float* __restrict__ fcb,
    float* __restrict__ out)
{
    extern __shared__ char smem[];
    _Float16* sWih0 = (_Float16*)(smem + S_WIH0);
    _Float16* sWhh0 = (_Float16*)(smem + S_WHH0);
    _Float16* sWih1 = (_Float16*)(smem + S_WIH1);
    _Float16* sWhh1 = (_Float16*)(smem + S_WHH1);
    _Float16* sFcW  = (_Float16*)(smem + S_FCW);
    float*    sB0   = (float*)(smem + S_B0);
    float*    sB1   = (float*)(smem + S_B1);
    float*    sFcb  = (float*)(smem + S_FCB);
    float*    sGat  = (float*)(smem + S_GATES);
    _Float16* sH0   = (_Float16*)(smem + S_H0);
    _Float16* sH1   = (_Float16*)(smem + S_H1);
    float*    sC0   = (float*)(smem + S_C0);
    float*    sC1   = (float*)(smem + S_C1);
    _Float16* sDin  = (_Float16*)(smem + S_DIN);

    const int tid  = threadIdx.x;
    const int lane = tid & 31;
    const int wave = tid >> 5;
    const int b0   = blockIdx.x * BTILE;

    // ---- stage encoder weights (f32 -> f16) & init state -------------------
    cvt_copy(sWih0, eWih0, G4 * NFEAT, tid);
    cvt_copy(sWhh0, eWhh0, G4 * HID, tid);
    cvt_copy(sWih1, eWih1, G4 * HID, tid);
    cvt_copy(sWhh1, eWhh1, G4 * HID, tid);
    cvt_copy(sFcW,  fcW,   NTGT * HID, tid);
    for (int i = tid; i < G4; i += NTHREADS) {
        sB0[i] = ebih0[i] + ebhh0[i];
        sB1[i] = ebih1[i] + ebhh1[i];
    }
    for (int i = tid; i < NTGT; i += NTHREADS) sFcb[i] = fcb[i];
    for (int i = tid; i < BTILE * HID; i += NTHREADS) {
        sH0[i] = (_Float16)0.0f; sH1[i] = (_Float16)0.0f;
        sC0[i] = 0.0f;           sC1[i] = 0.0f;
    }
    __syncthreads();

    // hoist this wave's invariant weight fragments into registers (112 VGPRs)
    WaveW W;
    load_wave_weights(W, sWih0, sWhh0, sWih1, sWhh1, sB0, sB1, lane, wave);

    // ---- encoder: fused 2-layer LSTM over the lookback window --------------
    for (int t = 0; t < LOOKBACK; ++t) {
        // stage x_t [32][32] into LDS f16; prefetch next timestep's rows
        for (int i = tid; i < BTILE * NFEAT; i += NTHREADS) {
            const int r = i >> 5, k = i & 31;
            const float* px = x + ((size_t)(b0 + r) * LOOKBACK + t) * NFEAT + k;
            sDin[i] = (_Float16)(*px);
            if (t + 1 < LOOKBACK) __builtin_prefetch(px + NFEAT, 0, 0);
        }
        __syncthreads();
        gemm_l0(W, sDin, sH0, sGat, lane, wave);
        __syncthreads();
        cell_update(sGat, sH0, sC0, tid);
        __syncthreads();
        gemm_l1(W, sH0, sH1, sGat, lane, wave);
        __syncthreads();
        cell_update(sGat, sH1, sC1, tid);
        __syncthreads();
    }

    // ---- swap LDS weights to decoder set -----------------------------------
    cvt_pad(sWih0, dWih0, G4, NTGT, NFEAT, tid);   // K padded 8 -> 32 with zeros
    cvt_copy(sWhh0, dWhh0, G4 * HID, tid);
    cvt_copy(sWih1, dWih1, G4 * HID, tid);
    cvt_copy(sWhh1, dWhh1, G4 * HID, tid);
    for (int i = tid; i < G4; i += NTHREADS) {
        sB0[i] = dbih0[i] + dbhh0[i];
        sB1[i] = dbih1[i] + dbhh1[i];
    }
    for (int i = tid; i < BTILE * NFEAT; i += NTHREADS) sDin[i] = (_Float16)0.0f; // dec_in0 = 0
    __syncthreads();

    // re-hoist decoder weight fragments
    load_wave_weights(W, sWih0, sWhh0, sWih1, sWhh1, sB0, sB1, lane, wave);

    // ---- decoder: autoregressive, feed pred back ----------------------------
    for (int s = 0; s < HORIZON; ++s) {
        gemm_l0(W, sDin, sH0, sGat, lane, wave);
        __syncthreads();
        cell_update(sGat, sH0, sC0, tid);
        __syncthreads();
        gemm_l1(W, sH0, sH1, sGat, lane, wave);
        __syncthreads();
        cell_update(sGat, sH1, sC1, tid);
        __syncthreads();
        // fc: pred[32][8] = h1 @ fcW^T + fcb ; one output element per thread
        {
            const int r = tid >> 3, c = tid & 7;
            float acc = sFcb[c];
            const _Float16* hr = sH1 + r * HID;
            const _Float16* wr = sFcW + c * HID;
#pragma unroll
            for (int k = 0; k < HID; ++k) acc = fmaf((float)hr[k], (float)wr[k], acc);
            out[((size_t)(b0 + r) * HORIZON + s) * NTGT + c] = acc;
            sDin[r * NFEAT + c] = (_Float16)acc;   // cols 8..31 stay zero
        }
        __syncthreads();
    }
}

// ---------------------------------------------------------------------------
extern "C" void kernel_launch(void* const* d_in, const int* in_sizes, int n_in,
                              void* d_out, int out_size, void* d_ws, size_t ws_size,
                              hipStream_t stream) {
    (void)in_sizes; (void)n_in; (void)out_size; (void)d_ws; (void)ws_size;

    const float* x     = (const float*)d_in[0];
    const float* eWih0 = (const float*)d_in[1];
    const float* eWhh0 = (const float*)d_in[2];
    const float* ebih0 = (const float*)d_in[3];
    const float* ebhh0 = (const float*)d_in[4];
    const float* eWih1 = (const float*)d_in[5];
    const float* eWhh1 = (const float*)d_in[6];
    const float* ebih1 = (const float*)d_in[7];
    const float* ebhh1 = (const float*)d_in[8];
    const float* dWih0 = (const float*)d_in[9];
    const float* dWhh0 = (const float*)d_in[10];
    const float* dbih0 = (const float*)d_in[11];
    const float* dbhh0 = (const float*)d_in[12];
    const float* dWih1 = (const float*)d_in[13];
    const float* dWhh1 = (const float*)d_in[14];
    const float* dbih1 = (const float*)d_in[15];
    const float* dbhh1 = (const float*)d_in[16];
    const float* fcW   = (const float*)d_in[17];
    const float* fcb   = (const float*)d_in[18];

    (void)hipFuncSetAttribute((const void*)seq2seq_lstm_kernel,
                              hipFuncAttributeMaxDynamicSharedMemorySize,
                              (int)S_TOTAL);

    dim3 grid(BATCH / BTILE);   // 128 workgroups, each owns 32 batch rows
    dim3 block(NTHREADS);       // 8 wave32 waves
    seq2seq_lstm_kernel<<<grid, block, S_TOTAL, stream>>>(
        x, eWih0, eWhh0, ebih0, ebhh0, eWih1, eWhh1, ebih1, ebhh1,
        dWih0, dWhh0, dbih0, dbhh0, dWih1, dWhh1, dbih1, dbhh1,
        fcW, fcb, (float*)d_out);
}
